// ReduceAttentionDecoder_81174881894987
// MI455X (gfx1250) — matrix operand order repaired and checked by reading, not compiled, not run on hardware
//
#include <hip/hip_runtime.h>
#include <hip/hip_bf16.h>

// ---- types for CDNA5 WMMA ----
typedef __bf16 bf16;
typedef __attribute__((ext_vector_type(16))) __bf16 v16bf;
typedef __attribute__((ext_vector_type(8)))  float  v8f;
typedef unsigned int uv4 __attribute__((ext_vector_type(4)));

#define HEADS 4
#define DHEAD 32
#define QLEN  4096   // 64*64
#define KLEN  1024   // 32*32

// ---------------- depthwise 3x3 pad1 (f32 -> f32, NCHW), used for x path ----------------
__global__ void dw3x3_kernel(const float* __restrict__ in, const float* __restrict__ w,
                             float* __restrict__ out, int C, int H, int W, int total) {
    int idx = blockIdx.x * blockDim.x + threadIdx.x;
    if (idx >= total) return;
    int x = idx % W; int t = idx / W;
    int y = t % H;   t /= H;
    int c = t % C;   int b = t / C;
    const float* ip = in + ((size_t)(b * C + c)) * H * W;
    const float* wp = w + c * 9;
    float s = 0.f;
#pragma unroll
    for (int dy = -1; dy <= 1; ++dy) {
        int yy = y + dy; if (yy < 0 || yy >= H) continue;
#pragma unroll
        for (int dx = -1; dx <= 1; ++dx) {
            int xx = x + dx; if (xx < 0 || xx >= W) continue;
            s += ip[yy * W + xx] * wp[(dy + 1) * 3 + (dx + 1)];
        }
    }
    out[idx] = s;
}

// ---------------- depthwise 3x3 on skip -> bf16 in [b][pixel][c] (A-matrix-ready) ----------------
__global__ void dw3x3_skip_kernel(const float* __restrict__ in, const float* __restrict__ w,
                                  bf16* __restrict__ out) {
    int idx = blockIdx.x * blockDim.x + threadIdx.x;   // total 8*128*4096
    if (idx >= 8 * 128 * QLEN) return;
    int x = idx & 63; int t = idx >> 6;
    int y = t & 63;   t >>= 6;
    int c = t & 127;  int b = t >> 7;
    const float* ip = in + ((size_t)(b * 128 + c)) * QLEN;
    const float* wp = w + c * 9;
    float s = 0.f;
#pragma unroll
    for (int dy = -1; dy <= 1; ++dy) {
        int yy = y + dy; if (yy < 0 || yy >= 64) continue;
#pragma unroll
        for (int dx = -1; dx <= 1; ++dx) {
            int xx = x + dx; if (xx < 0 || xx >= 64) continue;
            s += ip[yy * 64 + xx] * wp[(dy + 1) * 3 + (dx + 1)];
        }
    }
    out[((size_t)b * QLEN + (y * 64 + x)) * 128 + c] = (bf16)s;
}

// ---------------- f32 -> bf16 weight conversion (layout [o][c] kept: B-matrix-ready) ----------------
__global__ void wcvt_kernel(const float* __restrict__ src, bf16* __restrict__ dst, int n) {
    int i = blockIdx.x * blockDim.x + threadIdx.x;
    if (i < n) dst[i] = (bf16)src[i];
}

// ---------------- WMMA pointwise GEMM: q projection -> bf16 [b][p][128], scale folded ----------------
__global__ __launch_bounds__(128)
void gemm_q_kernel(const bf16* __restrict__ A, const bf16* __restrict__ W,
                   bf16* __restrict__ qout, float scale) {
    const int wave = threadIdx.x >> 5, lane = threadIdx.x & 31;
    const int half = lane >> 4, l16 = lane & 15;
    const int ptile = (blockIdx.x * 4 + wave) * 16;
    const int obase = blockIdx.y * 16;
    const int b     = blockIdx.z;
    const bf16* Ab = A + (size_t)b * QLEN * 128;
    v8f acc = {};
#pragma unroll
    for (int kk = 0; kk < 4; ++kk) {
        union { v16bf v; uv4 q[2]; } a;
        const bf16* ap = Ab + (size_t)(ptile + l16) * 128 + kk * 32 + half * 16;
        a.q[0] = *(const uv4*)(ap);
        a.q[1] = *(const uv4*)(ap + 8);
        v16bf bw = *(const v16bf*)(W + (size_t)(obase + l16) * 128 + kk * 32 + half * 16);
        acc = __builtin_amdgcn_wmma_f32_16x16x32_bf16(false, a.v, false, bw, (short)0, acc, false, false);
    }
#pragma unroll
    for (int e = 0; e < 8; ++e) {
        int p = ptile + half * 8 + e;
        qout[((size_t)b * QLEN + p) * 128 + obase + l16] = (bf16)(acc[e] * scale);
    }
}

// ---------------- WMMA pointwise GEMM: output projection -> f32 [b][o][p] (d_out layout) ----------------
__global__ __launch_bounds__(128)
void gemm_out_kernel(const bf16* __restrict__ A, const bf16* __restrict__ W,
                     float* __restrict__ out) {
    const int wave = threadIdx.x >> 5, lane = threadIdx.x & 31;
    const int half = lane >> 4, l16 = lane & 15;
    const int ptile = (blockIdx.x * 4 + wave) * 16;
    const int obase = blockIdx.y * 16;
    const int b     = blockIdx.z;
    const bf16* Ab = A + (size_t)b * QLEN * 128;
    v8f acc = {};
#pragma unroll
    for (int kk = 0; kk < 4; ++kk) {
        union { v16bf v; uv4 q[2]; } a;
        const bf16* ap = Ab + (size_t)(ptile + l16) * 128 + kk * 32 + half * 16;
        a.q[0] = *(const uv4*)(ap);
        a.q[1] = *(const uv4*)(ap + 8);
        v16bf bw = *(const v16bf*)(W + (size_t)(obase + l16) * 128 + kk * 32 + half * 16);
        acc = __builtin_amdgcn_wmma_f32_16x16x32_bf16(false, a.v, false, bw, (short)0, acc, false, false);
    }
#pragma unroll
    for (int e = 0; e < 8; ++e) {
        out[((size_t)b * 128 + obase + l16) * QLEN + ptile + half * 8 + e] = acc[e];
    }
}

// ---------------- pointwise 1x1 kv path (naive f32, tiny) ----------------
__global__ void pwkv_kernel(const float* __restrict__ t1, const float* __restrict__ wkv,
                            float* __restrict__ kvbuf) {
    int idx = blockIdx.x * blockDim.x + threadIdx.x;   // total 8*256*256
    if (idx >= 8 * 256 * 256) return;
    int p = idx % 256; int t = idx / 256;
    int o = t % 256;   int b = t / 256;
    const float* ip = t1 + (size_t)b * 256 * 256 + p;
    const float* wp = wkv + o * 256;
    float s = 0.f;
#pragma unroll 4
    for (int c = 0; c < 256; ++c) s += ip[(size_t)c * 256] * wp[c];
    kvbuf[((size_t)b * 256 + o) * 256 + p] = s;
}

// ---------------- bilinear resize (align_corners) 16->32 + bf16 pack (K rowmajor, V^T) ----------------
__global__ void resize_pack_kernel(const float* __restrict__ kvbuf,
                                   bf16* __restrict__ kb, bf16* __restrict__ vtb) {
    int idx = blockIdx.x * blockDim.x + threadIdx.x;   // total 8*128*1024
    if (idx >= 8 * 128 * KLEN) return;
    int op = idx % KLEN; int t = idx / KLEN;
    int ch = t % 128;    int b = t / 128;
    int oy = op >> 5, ox = op & 31;
    float ys = oy * (15.0f / 31.0f);
    float xs = ox * (15.0f / 31.0f);
    int y0 = (int)ys, x0 = (int)xs;
    int y1 = min(y0 + 1, 15), x1 = min(x0 + 1, 15);
    float wy = ys - y0, wx = xs - x0;

    const float* kp = kvbuf + ((size_t)b * 256 + ch) * 256;
    const float* vp = kvbuf + ((size_t)b * 256 + 128 + ch) * 256;
    float kr0 = kp[y0 * 16 + x0] * (1.f - wx) + kp[y0 * 16 + x1] * wx;
    float kr1 = kp[y1 * 16 + x0] * (1.f - wx) + kp[y1 * 16 + x1] * wx;
    float kv  = kr0 * (1.f - wy) + kr1 * wy;
    float vr0 = vp[y0 * 16 + x0] * (1.f - wx) + vp[y0 * 16 + x1] * wx;
    float vr1 = vp[y1 * 16 + x0] * (1.f - wx) + vp[y1 * 16 + x1] * wx;
    float vv  = vr0 * (1.f - wy) + vr1 * wy;

    int h = ch >> 5, d = ch & 31;
    kb [(((size_t)(b * HEADS + h) * KLEN) + op) * DHEAD + d] = (bf16)kv;   // [bh][kvpos][d]
    vtb[(((size_t)(b * HEADS + h) * DHEAD) + d) * KLEN + op] = (bf16)vv;   // [bh][d][kvpos]
}

// ---------------- flash attention core: 64-wide KV blocks, WMMA row-sum, bf16 out ----------------
__global__ __launch_bounds__(256)
void attn_kernel(const bf16* __restrict__ qb, const bf16* __restrict__ kb,
                 const bf16* __restrict__ vtb, bf16* __restrict__ ob) {
    __shared__ __align__(32) bf16 plds[8 * 16 * 64];   // 16KB: per-wave P repack slabs
    const int bh = blockIdx.y;                 // b*4+h
    const int b = bh >> 2, h = bh & 3;
    const int wave = threadIdx.x >> 5, lane = threadIdx.x & 31;
    const int half = lane >> 4, l16 = lane & 15;
    const int qtile = blockIdx.x * 128 + wave * 16;

    const bf16* K  = kb  + (size_t)bh * KLEN * DHEAD;
    const bf16* Vt = vtb + (size_t)bh * DHEAD * KLEN;
    bf16* myl = plds + wave * (16 * 64);

    // Resident A-matrix: 16x32 Q tile from [b][p][128] layout (d-contiguous run of 32).
    union { v16bf v; uv4 q[2]; } aq;
    {
        const bf16* qp = qb + ((size_t)b * QLEN + qtile + l16) * 128 + h * 32 + half * 16;
        aq.q[0] = *(const uv4*)(qp);
        aq.q[1] = *(const uv4*)(qp + 8);
    }

    v16bf onesB;
#pragma unroll
    for (int i = 0; i < 16; ++i) onesB[i] = (bf16)1.0f;

    float rowm[8], rowl[8];
    v8f acc0 = {}, acc1 = {};
#pragma unroll
    for (int e = 0; e < 8; ++e) { rowm[e] = -3.0e38f; rowl[e] = 0.0f; }

    for (int kblk = 0; kblk < KLEN / 64; ++kblk) {
        const int kvbase = kblk * 64;
        v8f z = {};
        v8f s[4];
#pragma unroll
        for (int j = 0; j < 4; ++j) {
            v16bf bk = *(const v16bf*)(K + (size_t)(kvbase + j * 16 + l16) * DHEAD + half * 16);
            s[j] = __builtin_amdgcn_wmma_f32_16x16x32_bf16(false, aq.v, false, bk, (short)0, z, false, false);
        }
        if (kblk + 1 < KLEN / 64) {   // global_prefetch_b8 of next K / V^T block
            __builtin_prefetch(K + (size_t)(kvbase + 64 + l16) * DHEAD, 0, 0);
            __builtin_prefetch(Vt + (size_t)l16 * KLEN + kvbase + 64, 0, 0);
        }

        // Online softmax: one 16-lane max reduction per row for the whole 64-wide block.
#pragma unroll
        for (int e = 0; e < 8; ++e) {
            float bm = fmaxf(fmaxf(s[0][e], s[1][e]), fmaxf(s[2][e], s[3][e]));
            bm = fmaxf(bm, __shfl_xor(bm, 1));
            bm = fmaxf(bm, __shfl_xor(bm, 2));
            bm = fmaxf(bm, __shfl_xor(bm, 4));
            bm = fmaxf(bm, __shfl_xor(bm, 8));
            float mnew = fmaxf(rowm[e], bm);
            float corr = __expf(rowm[e] - mnew);
            rowm[e] = mnew;
            rowl[e] *= corr;           // block sum added below via WMMA row-sum
            acc0[e] *= corr;
            acc1[e] *= corr;
            const int row = half * 8 + e;
#pragma unroll
            for (int j = 0; j < 4; ++j)
                myl[row * 64 + j * 16 + l16] = (bf16)__expf(s[j][e] - mnew);
        }
        __syncthreads();

        // Reload P as two A-matrices (kv cols 0..31, 32..63).
        union { v16bf v; uv4 q[2]; } pa0, pa1;
        {
            const bf16* pp = myl + l16 * 64 + half * 16;
            pa0.q[0] = *(const uv4*)(pp);
            pa0.q[1] = *(const uv4*)(pp + 8);
            pa1.q[0] = *(const uv4*)(pp + 32);
            pa1.q[1] = *(const uv4*)(pp + 40);
        }

        // Row-sums of P via WMMA against all-ones B (replaces 32 shuffle-adds).
        v8f rs = __builtin_amdgcn_wmma_f32_16x16x32_bf16(false, pa0.v, false, onesB, (short)0, z, false, false);
        rs     = __builtin_amdgcn_wmma_f32_16x16x32_bf16(false, pa1.v, false, onesB, (short)0, rs, false, false);
#pragma unroll
        for (int e = 0; e < 8; ++e) rowl[e] += rs[e];

        // O += P * V  (B operands from V^T rows: lane = output d column).
        v16bf bv;
        bv = *(const v16bf*)(Vt + (size_t)(l16)      * KLEN + kvbase + half * 16);
        acc0 = __builtin_amdgcn_wmma_f32_16x16x32_bf16(false, pa0.v, false, bv, (short)0, acc0, false, false);
        bv = *(const v16bf*)(Vt + (size_t)(l16)      * KLEN + kvbase + 32 + half * 16);
        acc0 = __builtin_amdgcn_wmma_f32_16x16x32_bf16(false, pa1.v, false, bv, (short)0, acc0, false, false);
        bv = *(const v16bf*)(Vt + (size_t)(16 + l16) * KLEN + kvbase + half * 16);
        acc1 = __builtin_amdgcn_wmma_f32_16x16x32_bf16(false, pa0.v, false, bv, (short)0, acc1, false, false);
        bv = *(const v16bf*)(Vt + (size_t)(16 + l16) * KLEN + kvbase + 32 + half * 16);
        acc1 = __builtin_amdgcn_wmma_f32_16x16x32_bf16(false, pa1.v, false, bv, (short)0, acc1, false, false);
    }

    // Normalize, write O as bf16 [b][p][c=h*32+d] (A-matrix-ready for output GEMM).
#pragma unroll
    for (int e = 0; e < 8; ++e) {
        int p = qtile + half * 8 + e;
        float inv = 1.0f / rowl[e];
        bf16* op = ob + ((size_t)b * QLEN + p) * 128 + h * 32;
        op[l16]      = (bf16)(acc0[e] * inv);
        op[16 + l16] = (bf16)(acc1[e] * inv);
    }
}

extern "C" void kernel_launch(void* const* d_in, const int* in_sizes, int n_in,
                              void* d_out, int out_size, void* d_ws, size_t ws_size,
                              hipStream_t stream) {
    const float* x    = (const float*)d_in[0];  // [8,256,16,16]
    const float* skip = (const float*)d_in[1];  // [8,128,64,64]
    const float* dwq  = (const float*)d_in[2];  // [128,1,3,3]
    const float* pwq  = (const float*)d_in[3];  // [128,128]
    const float* dwkv = (const float*)d_in[4];  // [256,1,3,3]
    const float* pwkv = (const float*)d_in[5];  // [256,256]
    const float* wout = (const float*)d_in[6];  // [128,128]
    float* out = (float*)d_out;                 // [8,128,64,64]

    char* ws = (char*)d_ws;
    bf16*  t2b   = (bf16*) (ws);                          // dw(skip) bf16 [b][p][c]:  8MB
    bf16*  qb2   = (bf16*) (ws + (size_t)( 8u << 20));    // Q bf16 [b][p][c]:         8MB
    bf16*  obbf  = (bf16*) (ws + (size_t)(16u << 20));    // attn out bf16 [b][p][c]:  8MB
    float* t1    = (float*)(ws + (size_t)(24u << 20));    // dw(x) f32:                2MB
    float* kvbuf = (float*)(ws + (size_t)(26u << 20));    // pw kv f32:                2MB
    bf16*  kb    = (bf16*) (ws + (size_t)(28u << 20));    // K bf16 [bh][kv][d]:       2MB
    bf16*  vtb   = (bf16*) (ws + (size_t)(30u << 20));    // V^T bf16 [bh][d][kv]:     2MB
    bf16*  wqb   = (bf16*) (ws + (size_t)(32u << 20));    // pw_q bf16 [o][c]:        32KB
    bf16*  wob   = (bf16*) (ws + (size_t)(33u << 20));    // w_out bf16 [o][c]:       32KB

    const float scale = 0.17677669529663687f;  // 32^-0.5

    {   // depthwise 3x3 on skip -> bf16 [b][p][c]
        int total = 8 * 128 * QLEN;
        dw3x3_skip_kernel<<<(total + 255) / 256, 256, 0, stream>>>(skip, dwq, t2b);
    }
    {   // weight conversions
        wcvt_kernel<<<(16384 + 255) / 256, 256, 0, stream>>>(pwq,  wqb, 16384);
        wcvt_kernel<<<(16384 + 255) / 256, 256, 0, stream>>>(wout, wob, 16384);
    }
    {   // q projection GEMM (WMMA), scale folded
        dim3 grid(QLEN / 64, 128 / 16, 8);
        gemm_q_kernel<<<grid, 128, 0, stream>>>(t2b, wqb, qb2, scale);
    }
    {   // depthwise 3x3 on x
        int total = 8 * 256 * 16 * 16;
        dw3x3_kernel<<<(total + 255) / 256, 256, 0, stream>>>(x, dwkv, t1, 256, 16, 16, total);
    }
    {   // kv pointwise (naive, tiny)
        int total = 8 * 256 * 256;
        pwkv_kernel<<<(total + 255) / 256, 256, 0, stream>>>(t1, pwkv, kvbuf);
    }
    {   // bilinear 16->32 + bf16 pack
        int total = 8 * 128 * KLEN;
        resize_pack_kernel<<<(total + 255) / 256, 256, 0, stream>>>(kvbuf, kb, vtb);
    }
    {   // WMMA flash attention
        dim3 grid(QLEN / 128, 32);
        attn_kernel<<<grid, 256, 0, stream>>>(qb2, kb, vtb, obbf);
    }
    {   // output projection GEMM (WMMA) -> d_out
        dim3 grid(QLEN / 64, 128 / 16, 8);
        gemm_out_kernel<<<grid, 128, 0, stream>>>(obbf, wob, out);
    }
}
